// Value_net_3779571220679
// MI455X (gfx1250) — matrix operand order repaired and checked by reading, not compiled, not run on hardware
//
#include <hip/hip_runtime.h>
#include <hip/hip_bf16.h>
#include <math.h>

typedef unsigned short u16;
typedef unsigned int   u32;
typedef __attribute__((ext_vector_type(16))) __bf16 v16bf;
typedef __attribute__((ext_vector_type(8)))  float  v8f;

#define HID    512
#define BATCH  64
#define TSTEPS 512
#define XKP    384   // 355 input features padded to a multiple of 32
#define NBLK   8     // persistent recurrence workgroups (HID/64)

__device__ __forceinline__ u16 f2bf(float f) {
  u32 u = __builtin_bit_cast(u32, f);
  u += 0x7FFFu + ((u >> 16) & 1u);          // round-to-nearest-even
  return (u16)(u >> 16);
}
__device__ __forceinline__ float bf2f(u16 h) {
  u32 u = ((u32)h) << 16;
  return __builtin_bit_cast(float, u);
}
// Load a 16x32 (A) or 32x16 (B) bf16 WMMA fragment: two 16-byte per-lane loads.
__device__ __forceinline__ v16bf load_frag(const u16* p0, const u16* p1) {
  union { uint4 u[2]; v16bf v; } f;
  f.u[0] = *(const uint4*)p0;
  f.u[1] = *(const uint4*)p1;
  return f.v;
}
__device__ __forceinline__ v8f wmma_bf16(v16bf a, v16bf b, v8f c) {
  return __builtin_amdgcn_wmma_f32_16x16x32_bf16(false, a, false, b, (short)0, c,
                                                 false, false);
}
__device__ __forceinline__ float sigmoidf_(float x) {
  return 1.0f / (1.0f + __expf(-x));
}

// ---------------- counter zeroing (per call: replay-safe barriers) -----------
__global__ void zero_ctr_kernel(u32* __restrict__ c, int n) {
  int i = blockIdx.x * blockDim.x + threadIdx.x;
  if (i < n) c[i] = 0u;
}

// ---------------- weight convert + pad to bf16 ----------------
__global__ void convert_pad_kernel(const float* __restrict__ src,
                                   u16* __restrict__ dst,
                                   int rows, int scols, int dcols) {
  int r = blockIdx.x;
  if (r >= rows) return;
  for (int c = threadIdx.x; c < dcols; c += blockDim.x) {
    float v = (c < scols) ? src[(size_t)r * scols + c] : 0.0f;
    dst[(size_t)r * dcols + c] = f2bf(v);
  }
}

// ---------------- build x = concat(time, week_e, hour_e, type_e, sp, tpp) ----
__global__ void build_x_kernel(const float* __restrict__ et,
                               const int*   __restrict__ ety,
                               const float* __restrict__ sp,
                               const float* __restrict__ tpp,
                               const float* __restrict__ wt,
                               const float* __restrict__ ht,
                               const float* __restrict__ ty,
                               u16* __restrict__ xbf) {
  int b = blockIdx.x;            // b = t*64 + n  (time-major rows)
  int t = b >> 6, n = b & 63;
  int si = n * TSTEPS + t;       // source layout is [N,T,...]
  float time = et[si];
  int widx = (int)fmodf(time, 7.0f);
  widx = widx < 0 ? 0 : (widx > 6 ? 6 : widx);
  int hidx = (int)(fmodf(time, 1.0f) * 24.0f);
  hidx = hidx < 0 ? 0 : (hidx > 23 ? 23 : hidx);
  int tyi = ety[si];
  tyi = tyi < 0 ? 0 : (tyi > 9 ? 9 : tyi);

  int c = threadIdx.x;           // 0..383
  float v;
  if (c == 0)        v = time;
  else if (c < 33)   v = wt[widx * 32 + (c - 1)];
  else if (c < 65)   v = ht[hidx * 32 + (c - 33)];
  else if (c < 97)   v = ty[tyi * 32 + (c - 65)];
  else if (c < 99)   v = sp[(size_t)si * 2 + (c - 97)];
  else if (c < 355)  v = tpp[(size_t)si * 256 + (c - 99)];
  else               v = 0.0f;
  xbf[(size_t)b * XKP + c] = f2bf(v);
}

// ---------------- persistent GRU recurrence --------------------------------
// 8 co-resident workgroups (one per 64 hidden cols), 512 threads = 16 waves.
// Wave w owns ONE 16x16 output tile: mt = w>>2 (batch rows), nt = w&3 (cols).
// fp32 hidden state for this wave's tile is carried in registers; the bf16
// hs[] record is only consumed as next step's A fragments + by the MLP.
__global__ __launch_bounds__(512) void gru_persistent_kernel(
    const u16* __restrict__ xbf,   // [T][64][XKP]
    u16*       __restrict__ hs,    // [T][64][HID]
    const u16* __restrict__ wih,   // [3*HID][XKP]
    const u16* __restrict__ whh,   // [3*HID][HID]
    const float* __restrict__ b_ih,
    const float* __restrict__ b_hh,
    u32* __restrict__ ctr) {       // [TSTEPS] arrival counters, pre-zeroed
  const int lane = threadIdx.x & 31;
  const int w    = threadIdx.x >> 5;  // 0..15
  const int mt   = w >> 2;            // 0..3
  const int nt   = w & 3;             // 0..3
  const int c0   = blockIdx.x * 64;   // hidden slice
  const int kh   = lane >> 4;         // lane half
  const int ncol = lane & 15;
  const int mrow = mt * 16 + ncol;    // A-fragment row (batch index 0..63)
  const int j    = c0 + nt * 16 + ncol;  // hidden unit (output column)

  // Loop-invariant: gate biases and weight row bases for this wave's tile.
  const float br_b  = b_ih[j] + b_hh[j];
  const float bz_b  = b_ih[HID + j] + b_hh[HID + j];
  const float bin_b = b_ih[2 * HID + j];
  const float bhn_b = b_hh[2 * HID + j];
  const u16* wihR = wih + (size_t)(0 * HID + j) * XKP;
  const u16* wihZ = wih + (size_t)(1 * HID + j) * XKP;
  const u16* wihN = wih + (size_t)(2 * HID + j) * XKP;
  const u16* whhR = whh + (size_t)(0 * HID + j) * HID;
  const u16* whhZ = whh + (size_t)(1 * HID + j) * HID;
  const u16* whhN = whh + (size_t)(2 * HID + j) * HID;

  v8f hprev = {};                     // fp32 carried hidden state (h0 = 0)

  for (int t = 0; t < TSTEPS; ++t) {
    asm volatile("" ::: "memory");    // no load CSE/hoist across time steps
    v8f accR = {}, accZ = {}, accGin = {}, accGhn = {};

    // ---- x part: K = 0..383 over W_ih ----
    const u16* xrow = xbf + ((size_t)t * BATCH + mrow) * XKP;
#pragma unroll 4
    for (int k0 = 0; k0 < XKP; k0 += 32) {
      v16bf a = load_frag(xrow + k0 + kh * 16, xrow + k0 + kh * 16 + 8);
      accR   = wmma_bf16(a, load_frag(wihR + k0 + kh * 16,
                                      wihR + k0 + kh * 16 + 8), accR);
      accZ   = wmma_bf16(a, load_frag(wihZ + k0 + kh * 16,
                                      wihZ + k0 + kh * 16 + 8), accZ);
      accGin = wmma_bf16(a, load_frag(wihN + k0 + kh * 16,
                                      wihN + k0 + kh * 16 + 8), accGin);
    }

    // ---- h part: K = 0..511 over W_hh (skip at t==0, h0 == 0) ----
    if (t > 0) {
      const u16* hrow = hs + ((size_t)(t - 1) * BATCH + mrow) * HID;
#pragma unroll 4
      for (int k0 = 0; k0 < HID; k0 += 32) {
        v16bf a = load_frag(hrow + k0 + kh * 16, hrow + k0 + kh * 16 + 8);
        accR   = wmma_bf16(a, load_frag(whhR + k0 + kh * 16,
                                        whhR + k0 + kh * 16 + 8), accR);
        accZ   = wmma_bf16(a, load_frag(whhZ + k0 + kh * 16,
                                        whhZ + k0 + kh * 16 + 8), accZ);
        accGhn = wmma_bf16(a, load_frag(whhN + k0 + kh * 16,
                                        whhN + k0 + kh * 16 + 8), accGhn);
      }
    }

    // ---- gates + state update (C/D layout: elem (m = vr + 8*kh, n = ncol)) --
#pragma unroll
    for (int vr = 0; vr < 8; ++vr) {
      const int m = mt * 16 + vr + 8 * kh;          // batch row 0..63
      float r  = sigmoidf_(accR[vr] + br_b);
      float z  = sigmoidf_(accZ[vr] + bz_b);
      float nv = tanhf(accGin[vr] + bin_b + r * (accGhn[vr] + bhn_b));
      float hn = (1.0f - z) * nv + z * hprev[vr];
      hprev[vr] = hn;                               // fp32 register carry
      hs[((size_t)t * BATCH + m) * HID + j] = f2bf(hn);
    }

    // ---- grid barrier: hs[t] fully visible before anyone reads it at t+1 ----
    __threadfence();                 // release my stores to device scope
    __syncthreads();
    if (threadIdx.x == 0) {
      atomicAdd(&ctr[t], 1u);
      while (__hip_atomic_load(&ctr[t], __ATOMIC_ACQUIRE,
                               __HIP_MEMORY_SCOPE_AGENT) < (u32)NBLK)
        __builtin_amdgcn_s_sleep(2);
    }
    __syncthreads();
    __threadfence();                 // acquire: drop stale lines before reads
  }
}

// ---------------- critic MLP: value = tanh(hs·W1^T + b1)·w2 + b2 -------------
// grid: (N*T)/16 blocks of 16 rows; 256 threads = 8 waves, 4 N-tiles each.
__global__ __launch_bounds__(256) void mlp_kernel(
    const u16* __restrict__ hs,    // [T*64][HID] bf16
    const u16* __restrict__ w1,    // [HID][HID] bf16
    const float* __restrict__ b1,
    const float* __restrict__ w2,  // [HID]
    const float* __restrict__ b2,  // [1]
    float* __restrict__ out) {     // [N][T]
  __shared__ float red[8][32][8];  // [wave][lane][vr] deterministic staging
  const int lane = threadIdx.x & 31;
  const int w    = threadIdx.x >> 5;
  const int kh   = lane >> 4;
  const int ncol = lane & 15;
  const size_t m0 = (size_t)blockIdx.x * 16;

  const u16* arow = hs + (m0 + ncol) * HID;
  float vacc[8] = {0, 0, 0, 0, 0, 0, 0, 0};

  for (int nt = w * 4; nt < w * 4 + 4; ++nt) {
    v8f acc = {};
    const u16* brow = w1 + (size_t)(nt * 16 + ncol) * HID;
#pragma unroll 4
    for (int k0 = 0; k0 < HID; k0 += 32) {
      v16bf a = load_frag(arow + k0 + kh * 16, arow + k0 + kh * 16 + 8);
      v16bf b = load_frag(brow + k0 + kh * 16, brow + k0 + kh * 16 + 8);
      acc = wmma_bf16(a, b, acc);
    }
    const int   col = nt * 16 + ncol;
    const float b1c = b1[col];
    const float w2c = w2[col];
#pragma unroll
    for (int vr = 0; vr < 8; ++vr)
      vacc[vr] += tanhf(acc[vr] + b1c) * w2c;
  }

#pragma unroll
  for (int vr = 0; vr < 8; ++vr) red[w][lane][vr] = vacc[vr];
  __syncthreads();

  if (threadIdx.x < 16) {
    const int r = threadIdx.x;                  // local row 0..15
    float s = 0.0f;
    for (int ww = 0; ww < 8; ++ww)
      for (int l = 0; l < 16; ++l)
        s += red[ww][(r >> 3) * 16 + l][r & 7]; // fixed order: deterministic
    const size_t g = m0 + r;                    // global row = t*64 + n
    const size_t t = g >> 6, n = g & 63;
    out[n * TSTEPS + t] = s + b2[0];
  }
}

extern "C" void kernel_launch(void* const* d_in, const int* in_sizes, int n_in,
                              void* d_out, int out_size, void* d_ws,
                              size_t ws_size, hipStream_t stream) {
  const float* event_times = (const float*)d_in[0];
  const int*   event_types = (const int*)d_in[1];
  const float* spatial     = (const float*)d_in[2];
  const float* tpp         = (const float*)d_in[3];
  const float* week_t      = (const float*)d_in[4];
  const float* hour_t      = (const float*)d_in[5];
  const float* type_t      = (const float*)d_in[6];
  const float* W_ih        = (const float*)d_in[7];
  const float* W_hh        = (const float*)d_in[8];
  const float* b_ih        = (const float*)d_in[9];
  const float* b_hh        = (const float*)d_in[10];
  const float* W1          = (const float*)d_in[11];
  const float* b1          = (const float*)d_in[12];
  const float* W2          = (const float*)d_in[13];
  const float* b2          = (const float*)d_in[14];
  float* out = (float*)d_out;

  char* ws = (char*)d_ws;
  size_t off = 0;
  u32* ctr = (u32*)(ws + off); off += 4096;                             // 2 KB used
  u16* xbf = (u16*)(ws + off); off += (size_t)TSTEPS * BATCH * XKP * 2; // 25.2 MB
  u16* wih = (u16*)(ws + off); off += (size_t)3 * HID * XKP * 2;        // 1.18 MB
  u16* whh = (u16*)(ws + off); off += (size_t)3 * HID * HID * 2;        // 1.57 MB
  u16* w1b = (u16*)(ws + off); off += (size_t)HID * HID * 2;            // 0.52 MB
  u16* hs  = (u16*)(ws + off);                                          // 33.5 MB

  zero_ctr_kernel<<<(TSTEPS + 127) / 128, 128, 0, stream>>>(ctr, TSTEPS);

  convert_pad_kernel<<<3 * HID, 128, 0, stream>>>(W_ih, wih, 3 * HID, 355, XKP);
  convert_pad_kernel<<<3 * HID, 128, 0, stream>>>(W_hh, whh, 3 * HID, HID, HID);
  convert_pad_kernel<<<HID, 128, 0, stream>>>(W1, w1b, HID, HID, HID);

  build_x_kernel<<<TSTEPS * BATCH, XKP, 0, stream>>>(
      event_times, event_types, spatial, tpp, week_t, hour_t, type_t, xbf);

  gru_persistent_kernel<<<NBLK, 512, 0, stream>>>(xbf, hs, wih, whh, b_ih,
                                                  b_hh, ctr);

  mlp_kernel<<<(TSTEPS * BATCH) / 16, 256, 0, stream>>>(hs, w1b, b1, W2, b2,
                                                        out);
}
